// SSMCausalConvBlock_49624052138253
// MI455X (gfx1250) — matrix operand order repaired and checked by reading, not compiled, not run on hardware
//
#include <hip/hip_runtime.h>
#include <cstddef>

typedef __attribute__((ext_vector_type(16))) _Float16 v16h;
typedef __attribute__((ext_vector_type(8)))  _Float16 h8;
typedef __attribute__((ext_vector_type(4)))  float    f32x4;
typedef __attribute__((ext_vector_type(8)))  float    v8f;
typedef unsigned int u32x4 __attribute__((ext_vector_type(4)));
typedef int          i32x4 __attribute__((ext_vector_type(4)));
typedef int          i32x8 __attribute__((ext_vector_type(8)));

#define BB 4
#define LL 8192
#define DD 128
#define NN 32
#define BL (BB * LL)            // 32768 rows
#define XCOLS (DD + 2 * NN)     // 192
#define CHUNK 128
#define NCHUNK (LL / CHUNK)     // 64
#define RT 4                    // row-tiles per block in GEMM kernels

#if __has_builtin(__builtin_amdgcn_tensor_load_to_lds) && \
    __has_builtin(__builtin_amdgcn_s_wait_tensorcnt)
#define USE_TDM 1
#endif

// ---------------------------------------------------------------------------
// Stage a contiguous block of f32 from global into LDS.
// TDM path: 1-D tensor descriptor (data_size=4B, tile_dim0=ne, 1 row),
// issued by wave 0, completion via s_wait_tensorcnt. Caller must
// __syncthreads() afterwards. Fallback: cooperative copy.
// ---------------------------------------------------------------------------
__device__ __forceinline__ void stage_f32(float* dst, const float* __restrict__ src,
                                          int ne, int tid) {
#ifdef USE_TDM
  if (tid < 32) {
    unsigned lds = (unsigned)(size_t)dst;     // low 32 bits = LDS byte address
    unsigned long long ga = (unsigned long long)(size_t)src;
    u32x4 g0;
    g0[0] = 1u;                                        // count = 1 (valid D#)
    g0[1] = lds;                                       // lds_addr
    g0[2] = (unsigned)(ga & 0xffffffffu);              // global_addr[31:0]
    g0[3] = (unsigned)((ga >> 32) & 0x1ffffffu) | (2u << 30);  // [56:32]|type=2
    i32x8 g1;
    g1[0] = 0x20000;                                   // data_size = 4B
    g1[1] = (int)((ne & 0xffff) << 16);                // tensor_dim0 lo16
    g1[2] = (int)((((unsigned)ne >> 16) & 0xffffu) | (1u << 16)); // hi16 | dim1=1
    g1[3] = (int)((ne & 0xffff) << 16);                // tile_dim0 = ne
    g1[4] = 1;                                         // tile_dim1 = 1
    g1[5] = ne;                                        // tensor_dim0_stride
    g1[6] = 0;
    g1[7] = 0;
    i32x4 z4 = {0, 0, 0, 0};
#if __clang_major__ >= 23
    i32x8 z8 = {0, 0, 0, 0, 0, 0, 0, 0};
    __builtin_amdgcn_tensor_load_to_lds(g0, g1, z4, z4, z8, 0);
#else
    __builtin_amdgcn_tensor_load_to_lds(g0, g1, z4, z4, 0);
#endif
    __builtin_amdgcn_s_wait_tensorcnt(0);
  }
#else
  for (int i = tid; i < ne; i += 128) dst[i] = src[i];
#endif
}

// ---------------------------------------------------------------------------
// WMMA fragment helpers (CDNA5 wave32 layouts, f16 A/B, f32 C/D).
// A (16x32, MxK): lanes 0-15 rows M=0..15 with K = {0..7,16..23}; lanes 16-31
//                 same rows with K = {8..15,24..31} -> two contiguous runs.
// B (32x16, KxN): lane&15 = column N; lanes>=16 carry K={8..15,24..31}.
// C/D (16x16):    lane&15 = N; VGPR r -> M = r + 8*(lane>=16).
// ---------------------------------------------------------------------------
__device__ __forceinline__ v16h load_a_frag(const float* src, int ld, int lane, int k0) {
  int m  = lane & 15;
  int kb = (lane >> 4) ? 8 : 0;
  const float* row = src + m * ld + k0 + kb;
  f32x4 q0 = *(const f32x4*)(row);        // K = kb+0..3
  f32x4 q1 = *(const f32x4*)(row + 4);    // K = kb+4..7
  f32x4 q2 = *(const f32x4*)(row + 16);   // K = kb+16..19
  f32x4 q3 = *(const f32x4*)(row + 20);   // K = kb+20..23
  v16h a;
#pragma unroll
  for (int i = 0; i < 4; ++i) {
    a[i]      = (_Float16)q0[i];
    a[4 + i]  = (_Float16)q1[i];
    a[8 + i]  = (_Float16)q2[i];
    a[12 + i] = (_Float16)q3[i];
  }
  return a;
}

__device__ __forceinline__ v16h load_a_frag_h(const _Float16* src, int ld, int lane, int k0) {
  int m  = lane & 15;
  int kb = (lane >> 4) ? 8 : 0;
  const _Float16* row = src + m * ld + k0 + kb;
  h8 u0 = *(const h8*)(row);        // K = kb+0..7
  h8 u1 = *(const h8*)(row + 16);   // K = kb+16..23
  v16h a;
#pragma unroll
  for (int i = 0; i < 8; ++i) { a[i] = u0[i]; a[8 + i] = u1[i]; }
  return a;
}

// Strided gather of a B fragment from a row-major f32 global weight matrix.
__device__ __forceinline__ v16h load_b_frag(const float* W, int ldn, int lane, int k0, int n0) {
  v16h b;
  int n  = n0 + (lane & 15);
  int kb = (lane >> 4) ? 8 : 0;
#pragma unroll
  for (int h = 0; h < 16; ++h) {
    b[h] = (_Float16)W[(size_t)(k0 + kb + (h & 7) + ((h >> 3) << 4)) * ldn + n];
  }
  return b;
}

// ---------------------------------------------------------------------------
// Swizzled (fragment-major) f16 weight staging: fragment (nt, kt) occupies
// 512 consecutive halves; within a fragment lane's 16 halves are contiguous
// (32 bytes) -> a whole B fragment is one v16h LDS load (2x ds_load_b128).
// ---------------------------------------------------------------------------
__device__ __forceinline__ void stage_weights_sw(_Float16* dst, const float* __restrict__ W,
                                                 int Ktot, int Ntot, int tid) {
  int nKt = Ktot >> 5;
  int total = Ktot * Ntot;
  for (int i = tid; i < total; i += 128) {
    int frag = i >> 9;            // fragment index = nt*nKt + kt
    int rem  = i & 511;
    int lane = rem >> 4;
    int h    = rem & 15;
    int nt   = frag / nKt;
    int kt   = frag - nt * nKt;
    int n    = nt * 16 + (lane & 15);
    int k    = kt * 32 + ((lane >> 4) << 3) + (h & 7) + ((h >> 3) << 4);
    dst[i] = (_Float16)W[(size_t)k * Ntot + n];
  }
}

__device__ __forceinline__ v16h load_b_sw(const _Float16* sW, int nKt, int kt, int nt, int lane) {
  return *(const v16h*)(sW + (((nt * nKt + kt) << 9) + (lane << 4)));
}

__device__ __forceinline__ v8f wmma_f16(v16h a, v16h b, v8f c) {
  return __builtin_amdgcn_wmma_f32_16x16x32_f16(false, a, false, b, (short)0, c,
                                                false, false);
}

// ---------------------------------------------------------------------------
// k0: A = -exp(A_log)
// ---------------------------------------------------------------------------
__global__ __launch_bounds__(256) void k0_aneg(const float* __restrict__ A_log,
                                               float* __restrict__ Aneg) {
  int i = blockIdx.x * 256 + threadIdx.x;
  if (i < DD * NN) Aneg[i] = -__expf(A_log[i]);
}

// ---------------------------------------------------------------------------
// k1: x_dbl = x @ W_x ; P = Bm*Cm ; delta = softplus(delta_raw @ W_dt + b_dt)
// 64 rows (4 row-tiles) per block; W_x staged once per block, swizzled f16.
// ---------------------------------------------------------------------------
__global__ __launch_bounds__(128) void k1_frontend(
    const float* __restrict__ x, const float* __restrict__ Wx,
    const float* __restrict__ Wdt, const float* __restrict__ bdt,
    float* __restrict__ delta, float* __restrict__ P) {
  __shared__ __align__(32) _Float16 sWx[DD * XCOLS];   // 48 KB, swizzled
  __shared__ __align__(16) float    sX[16 * DD];       // 8 KB (raw f32: TDM)
  __shared__ __align__(16) _Float16 sDraw[16 * DD];    // 4 KB
  __shared__ float                  sBC[16 * 2 * NN];  // 4 KB

  int tid  = threadIdx.x;
  int lane = tid & 31;
  int wave = tid >> 5;

  stage_weights_sw(sWx, Wx, DD, XCOLS, tid);

  for (int t = 0; t < RT; ++t) {
    int r0 = blockIdx.x * (16 * RT) + t * 16;
    stage_f32(sX, x + (size_t)r0 * DD, 16 * DD, tid);
    if (r0 + 32 <= BL)   // prefetch next row-tile behind this tile's compute
      __builtin_prefetch((const char*)(x + (size_t)(r0 + 16) * DD) + tid * 64, 0, 0);
    __syncthreads();

    // GEMM1: 16 x 192, K = 128
    for (int ct = wave; ct < XCOLS / 16; ct += 4) {
      v8f acc = {};
#pragma unroll
      for (int kt = 0; kt < 4; ++kt) {
        v16h a = load_a_frag(sX, DD, lane, kt * 32);
        v16h b = load_b_sw(sWx, 4, kt, ct, lane);
        acc = wmma_f16(a, b, acc);
      }
      int n  = ct * 16 + (lane & 15);
      int mb = (lane >> 4) * 8;
      if (ct < DD / 16) {            // wave-uniform: delta_raw columns
#pragma unroll
        for (int r = 0; r < 8; ++r)
          sDraw[(mb + r) * DD + n] = (_Float16)acc[r];
      } else {                       // wave-uniform: Bm/Cm columns
#pragma unroll
        for (int r = 0; r < 8; ++r)
          sBC[(mb + r) * (2 * NN) + (n - DD)] = acc[r];
      }
    }
    __syncthreads();

    // P = Bm * Cm
    for (int i = tid; i < 16 * NN; i += 128) {
      int m = i >> 5, n = i & 31;
      P[(size_t)(r0 + m) * NN + n] =
          sBC[m * (2 * NN) + n] * sBC[m * (2 * NN) + NN + n];
    }

    // GEMM2: 16 x 128, K = 128, softplus epilogue
    for (int ct = wave; ct < DD / 16; ct += 4) {
      v8f acc = {};
#pragma unroll
      for (int kt = 0; kt < 4; ++kt) {
        v16h a = load_a_frag_h(sDraw, DD, lane, kt * 32);
        v16h b = load_b_frag(Wdt, DD, lane, kt * 32, ct * 16);
        acc = wmma_f16(a, b, acc);
      }
      int n  = ct * 16 + (lane & 15);
      int mb = (lane >> 4) * 8;
#pragma unroll
      for (int r = 0; r < 8; ++r) {
        float v = acc[r] + bdt[n];
        v = (v > 20.f) ? v : log1pf(__expf(v));   // softplus
        delta[(size_t)(r0 + mb + r) * DD + n] = v;
      }
    }
    __syncthreads();
  }
}

// ---------------------------------------------------------------------------
// k2a: per-(b,chunk) exclusive prefix of delta over L; lane = d (coalesced).
// ---------------------------------------------------------------------------
__global__ __launch_bounds__(128) void k2a_chunkscan(
    const float* __restrict__ delta, float* __restrict__ Sloc,
    float* __restrict__ bsum) {
  int b = blockIdx.x / NCHUNK;
  int c = blockIdx.x % NCHUNK;
  int d = threadIdx.x;
  size_t base = ((size_t)b * LL + (size_t)c * CHUNK) * DD + d;
  float run = 0.f;
  for (int i = 0; i < CHUNK; ++i) {
    float v = delta[base + (size_t)i * DD];
    Sloc[base + (size_t)i * DD] = run;
    run += v;
  }
  bsum[((size_t)b * NCHUNK + c) * DD + d] = run;
}

// k2b: scan of chunk sums (exclusive) per (b, d).
__global__ __launch_bounds__(128) void k2b_blockscan(
    const float* __restrict__ bsum, float* __restrict__ boff) {
  int b = blockIdx.x;
  int d = threadIdx.x;
  float run = 0.f;
  for (int c = 0; c < NCHUNK; ++c) {
    size_t idx = ((size_t)b * NCHUNK + c) * DD + d;
    boff[idx] = run;
    run += bsum[idx];
  }
}

// ---------------------------------------------------------------------------
// k3: h = delta*hprev*(1+Dp)*sum_n P(b,l,n)*exp(A(d,n)*S(b,l,d))
// ---------------------------------------------------------------------------
__global__ __launch_bounds__(256) void k3_state(
    const float* __restrict__ delta, const float* __restrict__ Sloc,
    const float* __restrict__ boff, const float* __restrict__ P,
    const float* __restrict__ Aneg, const float* __restrict__ hprev,
    const float* __restrict__ Dp, float* __restrict__ hout) {
  size_t idx = (size_t)blockIdx.x * 256 + threadIdx.x;   // over B*L*D
  int d = (int)(idx & (DD - 1));
  size_t bl = idx >> 7;                                  // b*L + l
  int b = (int)(bl >> 13);
  int l = (int)(bl & (LL - 1));
  int c = l >> 7;
  float S = Sloc[idx] + boff[((size_t)b * NCHUNK + c) * DD + d];
  const float* Pn = P + bl * NN;
  const float* An = Aneg + (size_t)d * NN;
  float acc = 0.f;
#pragma unroll
  for (int n = 0; n < NN; ++n)
    acc = fmaf(Pn[n], __expf(An[n] * S), acc);
  float h = delta[idx] * hprev[idx] * acc;
  hout[idx] = h * (1.f + Dp[d]);
}

// ---------------------------------------------------------------------------
// k4a: ssm = relu(h @ W_f + b_f); W_f staged once per block, swizzled f16.
// ---------------------------------------------------------------------------
__global__ __launch_bounds__(128) void k4a_ssm(
    const float* __restrict__ h, const float* __restrict__ Wf,
    const float* __restrict__ bf, float* __restrict__ ssm) {
  __shared__ __align__(32) _Float16 sWf[DD * DD];   // 32 KB, swizzled
  __shared__ __align__(16) float    sH[16 * DD];    // 8 KB

  int tid  = threadIdx.x;
  int lane = tid & 31;
  int wave = tid >> 5;

  stage_weights_sw(sWf, Wf, DD, DD, tid);

  for (int t = 0; t < RT; ++t) {
    int r0 = blockIdx.x * (16 * RT) + t * 16;
    stage_f32(sH, h + (size_t)r0 * DD, 16 * DD, tid);
    if (r0 + 32 <= BL)
      __builtin_prefetch((const char*)(h + (size_t)(r0 + 16) * DD) + tid * 64, 0, 0);
    __syncthreads();

    for (int ct = wave; ct < DD / 16; ct += 4) {
      v8f acc = {};
#pragma unroll
      for (int kt = 0; kt < 4; ++kt) {
        v16h a = load_a_frag(sH, DD, lane, kt * 32);
        v16h b = load_b_sw(sWf, 4, kt, ct, lane);
        acc = wmma_f16(a, b, acc);
      }
      int n  = ct * 16 + (lane & 15);
      int mb = (lane >> 4) * 8;
#pragma unroll
      for (int r = 0; r < 8; ++r) {
        float v = acc[r] + bf[n];
        ssm[(size_t)(r0 + mb + r) * DD + n] = v > 0.f ? v : 0.f;
      }
    }
    __syncthreads();
  }
}

// ---------------------------------------------------------------------------
// k4b: causal conv (k=3) as 3 shifted WMMA GEMMs + relu + fuse with ssm.
// Stages the 18-row (l0-2 .. l0+15) halo tile via TDM; first tile of each
// batch zero-fills the 2 halo rows. L=8192 is 64-row aligned, so no tile
// crosses a batch boundary.
// ---------------------------------------------------------------------------
__global__ __launch_bounds__(128) void k4b_conv(
    const float* __restrict__ ssm, const float* __restrict__ convw,
    const float* __restrict__ convb, float* __restrict__ out) {
  __shared__ __align__(16) float sS[18 * DD];   // 9 KB

  int tid  = threadIdx.x;
  int lane = tid & 31;
  int wave = tid >> 5;

  for (int t = 0; t < RT; ++t) {
    int r0 = blockIdx.x * (16 * RT) + t * 16;
    int l0 = r0 & (LL - 1);
    if (l0 == 0) {
      for (int i = tid; i < 2 * DD; i += 128) sS[i] = 0.f;  // left padding
      stage_f32(sS + 2 * DD, ssm + (size_t)r0 * DD, 16 * DD, tid);
    } else {
      stage_f32(sS, ssm + (size_t)(r0 - 2) * DD, 18 * DD, tid);
    }
    if (r0 + 32 <= BL)
      __builtin_prefetch((const char*)(ssm + (size_t)(r0 + 14) * DD) + tid * 64, 0, 0);
    __syncthreads();

    for (int ct = wave; ct < DD / 16; ct += 4) {
      v8f acc = {};
#pragma unroll
      for (int w = 0; w < 3; ++w) {
        const float* Wc = convw + (size_t)w * DD * DD;   // conv_w[w][k][n]
#pragma unroll
        for (int kt = 0; kt < 4; ++kt) {
          // output row m uses staged input row (m + w)
          v16h a = load_a_frag(sS + w * DD, DD, lane, kt * 32);
          v16h b = load_b_frag(Wc, DD, lane, kt * 32, ct * 16);
          acc = wmma_f16(a, b, acc);
        }
      }
      int n  = ct * 16 + (lane & 15);
      int mb = (lane >> 4) * 8;
#pragma unroll
      for (int r = 0; r < 8; ++r) {
        int m = mb + r;
        float c = acc[r] + convb[n];
        c = c > 0.f ? c : 0.f;                           // relu(conv)
        float s = sS[(m + 2) * DD + n];                  // ssm(l)
        out[(size_t)(r0 + m) * DD + n] = s * c;          // fused output
      }
    }
    __syncthreads();
  }
}

// ---------------------------------------------------------------------------
extern "C" void kernel_launch(void* const* d_in, const int* in_sizes, int n_in,
                              void* d_out, int out_size, void* d_ws,
                              size_t ws_size, hipStream_t stream) {
  const float* x     = (const float*)d_in[0];
  const float* hprev = (const float*)d_in[1];
  const float* A_log = (const float*)d_in[2];
  const float* Dp    = (const float*)d_in[3];
  const float* Wx    = (const float*)d_in[4];
  const float* Wdt   = (const float*)d_in[5];
  const float* bdt   = (const float*)d_in[6];
  const float* convw = (const float*)d_in[7];
  const float* convb = (const float*)d_in[8];
  const float* Wf    = (const float*)d_in[9];
  const float* bf    = (const float*)d_in[10];

  float* out  = (float*)d_out;                  // fused_output (B,L,D)
  float* hout = out + (size_t)BL * DD;          // h_t           (B,L,D)

  float* ws    = (float*)d_ws;
  float* delta = ws;                            // BL*DD
  float* P     = delta + (size_t)BL * DD;       // BL*NN
  float* Sloc  = P + (size_t)BL * NN;           // BL*DD
  float* bsum  = Sloc + (size_t)BL * DD;        // BB*NCHUNK*DD
  float* boff  = bsum + (size_t)BB * NCHUNK * DD;
  float* Aneg  = boff + (size_t)BB * NCHUNK * DD;  // DD*NN
  float* ssm   = Aneg + (size_t)DD * NN;        // BL*DD

  int gemmBlocks = BL / (16 * RT);              // 512

  k0_aneg<<<(DD * NN + 255) / 256, 256, 0, stream>>>(A_log, Aneg);
  k1_frontend<<<gemmBlocks, 128, 0, stream>>>(x, Wx, Wdt, bdt, delta, P);
  k2a_chunkscan<<<BB * NCHUNK, 128, 0, stream>>>(delta, Sloc, bsum);
  k2b_blockscan<<<BB, 128, 0, stream>>>(bsum, boff);
  k3_state<<<(int)(((size_t)BL * DD) / 256), 256, 0, stream>>>(
      delta, Sloc, boff, P, Aneg, hprev, Dp, hout);
  k4a_ssm<<<gemmBlocks, 128, 0, stream>>>(hout, Wf, bf, ssm);
  k4b_conv<<<gemmBlocks, 128, 0, stream>>>(ssm, convw, convb, out);
}
